// ComplementConstraintCombined_7825430413734
// MI455X (gfx1250) — matrix operand order repaired and checked by reading, not compiled, not run on hardware
//
#include <hip/hip_runtime.h>
#include <hip/hip_bf16.h>
#include <math.h>

typedef __attribute__((ext_vector_type(16))) __bf16 v16bf;
typedef __attribute__((ext_vector_type(8)))  __bf16 v8bf;
typedef __attribute__((ext_vector_type(8)))  float  v8f;

#define BDIM 8192
#define CDIM 1000
#define DDIM 2048

#define BM 128
#define BN 128
#define BK 32
#define LDT 40   // padded LDS row stride in bf16 elements (80 bytes)

__device__ __forceinline__ void split_bf16(float f, __bf16& h, __bf16& l) {
    h = (__bf16)f;
    l = (__bf16)(f - (float)h);
}

__global__ __launch_bounds__(256)
void gemm_bias_bf16x3_kernel(const float* __restrict__ x,
                             const float* __restrict__ W,
                             const float* __restrict__ bias,
                             float* __restrict__ out)
{
    __shared__ __align__(16) __bf16 Ash[BM * LDT];
    __shared__ __align__(16) __bf16 Asl[BM * LDT];
    __shared__ __align__(16) __bf16 Bsh[BN * LDT];
    __shared__ __align__(16) __bf16 Bsl[BN * LDT];

    const int tid   = threadIdx.x;
    const int lane  = tid & 31;
    const int wave  = tid >> 5;      // 0..7
    const int waveM = wave & 3;      // 4 waves along M: 32 rows each
    const int waveN = wave >> 2;     // 2 waves along N: 64 cols each
    const int mBase = blockIdx.x * BM;
    const int nBase = blockIdx.y * BN;

    const int ln16 = lane & 15;
    const int hi16 = lane >> 4;      // 0 for lanes 0-15, 1 for lanes 16-31

    v8f acc[2][4] = {};

    for (int k0 = 0; k0 < DDIM; k0 += BK) {
        // ---- stage A tile (BM x BK f32) as split bf16 hi/lo into LDS ----
        #pragma unroll
        for (int i = 0; i < 4; ++i) {
            const int seg = i * 256 + tid;        // 1024 float4 segments
            const int row = seg >> 3;             // 8 segments per 32-wide row
            const int ks  = (seg & 7) << 2;
            const float4 v = *(const float4*)(x + (row + mBase) * DDIM + k0 + ks);
            const float f[4] = { v.x, v.y, v.z, v.w };
            #pragma unroll
            for (int j = 0; j < 4; ++j) {
                __bf16 h, l;
                split_bf16(f[j], h, l);
                Ash[row * LDT + ks + j] = h;
                Asl[row * LDT + ks + j] = l;
            }
        }
        // ---- stage B tile (BK x BN f32 of W), transposed to [N][K] bf16 ----
        #pragma unroll
        for (int i = 0; i < 4; ++i) {
            const int seg = i * 256 + tid;
            const int kk  = seg >> 5;             // 32 segments per k-row
            const int cj  = (seg & 31) << 2;      // local col 0..124
            const int col = nBase + cj;
            float f[4];
            if (col + 3 < CDIM) {
                const float4 v = *(const float4*)(W + (k0 + kk) * CDIM + col);
                f[0] = v.x; f[1] = v.y; f[2] = v.z; f[3] = v.w;
            } else {
                #pragma unroll
                for (int j = 0; j < 4; ++j)
                    f[j] = (col + j < CDIM) ? W[(k0 + kk) * CDIM + col + j] : 0.0f;
            }
            #pragma unroll
            for (int j = 0; j < 4; ++j) {
                __bf16 h, l;
                split_bf16(f[j], h, l);
                Bsh[(cj + j) * LDT + kk] = h;     // transposed store
                Bsl[(cj + j) * LDT + kk] = l;
            }
        }
        __syncthreads();

        // ---- A fragments (16x32 bf16 per 16-row tile), ISA 7.12.2 layout ----
        // lanes 0-15: K chunks {0..7} and {16..23}; lanes 16-31: {8..15},{24..31}
        v16bf ah[2], al[2];
        const int ac0 = hi16 ? 8 : 0;
        const int ac1 = hi16 ? 24 : 16;
        #pragma unroll
        for (int mt = 0; mt < 2; ++mt) {
            const int row = waveM * 32 + mt * 16 + ln16;
            const v8bf h0 = *(const v8bf*)&Ash[row * LDT + ac0];
            const v8bf h1 = *(const v8bf*)&Ash[row * LDT + ac1];
            const v8bf l0 = *(const v8bf*)&Asl[row * LDT + ac0];
            const v8bf l1 = *(const v8bf*)&Asl[row * LDT + ac1];
            #pragma unroll
            for (int e = 0; e < 8; ++e) {
                ah[mt][e] = h0[e]; ah[mt][8 + e] = h1[e];
                al[mt][e] = l0[e]; al[mt][8 + e] = l1[e];
            }
        }
        // ---- B fragments (32x16): lane col = lane%16, K chunks {0..7},{16..23} / {8..15},{24..31} ----
        v16bf bh[4], bl[4];
        const int bk0 = hi16 * 8;
        #pragma unroll
        for (int nt = 0; nt < 4; ++nt) {
            const int col = waveN * 64 + nt * 16 + ln16;   // local col in tile
            const v8bf h0 = *(const v8bf*)&Bsh[col * LDT + bk0];
            const v8bf h1 = *(const v8bf*)&Bsh[col * LDT + bk0 + 16];
            const v8bf l0 = *(const v8bf*)&Bsl[col * LDT + bk0];
            const v8bf l1 = *(const v8bf*)&Bsl[col * LDT + bk0 + 16];
            #pragma unroll
            for (int e = 0; e < 8; ++e) {
                bh[nt][e] = h0[e]; bh[nt][8 + e] = h1[e];
                bl[nt][e] = l0[e]; bl[nt][8 + e] = l1[e];
            }
        }

        // ---- error-compensated bf16x3 WMMA: hi*hi + hi*lo + lo*hi ----
        #pragma unroll
        for (int mt = 0; mt < 2; ++mt) {
            #pragma unroll
            for (int nt = 0; nt < 4; ++nt) {
                acc[mt][nt] = __builtin_amdgcn_wmma_f32_16x16x32_bf16(
                    false, ah[mt], false, bh[nt], (short)0, acc[mt][nt], false, false);
                acc[mt][nt] = __builtin_amdgcn_wmma_f32_16x16x32_bf16(
                    false, ah[mt], false, bl[nt], (short)0, acc[mt][nt], false, false);
                acc[mt][nt] = __builtin_amdgcn_wmma_f32_16x16x32_bf16(
                    false, al[mt], false, bh[nt], (short)0, acc[mt][nt], false, false);
            }
        }
        __syncthreads();
    }

    // ---- epilogue: add bias, store. C/D layout: VGPR i -> M = i (+8 for hi lanes), N = lane%16
    #pragma unroll
    for (int nt = 0; nt < 4; ++nt) {
        const int col = nBase + waveN * 64 + nt * 16 + ln16;
        if (col < CDIM) {
            const float bv = bias[col];
            #pragma unroll
            for (int mt = 0; mt < 2; ++mt) {
                const int r0 = mBase + waveM * 32 + mt * 16 + (hi16 ? 8 : 0);
                #pragma unroll
                for (int e = 0; e < 8; ++e)
                    out[(r0 + e) * CDIM + col] = acc[mt][nt][e] + bv;
            }
        }
    }
}

__global__ __launch_bounds__(256)
void loo_lse_kernel(float* __restrict__ out)
{
    const int row = blockIdx.x;
    const int t   = threadIdx.x;
    const int wid = t >> 5;
    const int lid = t & 31;
    __shared__ float red[8];

    float v[4];
    int   cc[4];
    float mx = -__builtin_inff();
    #pragma unroll
    for (int i = 0; i < 4; ++i) {
        const int c = t + i * 256;
        cc[i] = c;
        if (c < CDIM) { v[i] = out[row * CDIM + c]; mx = fmaxf(mx, v[i]); }
        else          { v[i] = -__builtin_inff(); }
    }
    // block max
    #pragma unroll
    for (int off = 16; off > 0; off >>= 1)
        mx = fmaxf(mx, __shfl_xor(mx, off, 32));
    if (lid == 0) red[wid] = mx;
    __syncthreads();
    if (wid == 0) {
        float m = (lid < 8) ? red[lid] : -__builtin_inff();
        #pragma unroll
        for (int off = 4; off > 0; off >>= 1)
            m = fmaxf(m, __shfl_xor(m, off, 32));
        if (lid == 0) red[0] = m;
    }
    __syncthreads();
    mx = red[0];
    __syncthreads();

    // block sum of exp
    float s = 0.0f;
    #pragma unroll
    for (int i = 0; i < 4; ++i)
        if (cc[i] < CDIM) s += expf(v[i] - mx);
    #pragma unroll
    for (int off = 16; off > 0; off >>= 1)
        s += __shfl_xor(s, off, 32);
    if (lid == 0) red[wid] = s;
    __syncthreads();
    if (wid == 0) {
        float s2 = (lid < 8) ? red[lid] : 0.0f;
        #pragma unroll
        for (int off = 4; off > 0; off >>= 1)
            s2 += __shfl_xor(s2, off, 32);
        if (lid == 0) red[0] = s2;
    }
    __syncthreads();
    s = red[0];

    const float lse = mx + logf(s);
    #pragma unroll
    for (int i = 0; i < 4; ++i) {
        if (cc[i] < CDIM) {
            const float d = v[i] - lse;             // out - lse  (< 0)
            out[row * CDIM + cc[i]] = d - log1pf(-expf(d));
        }
    }
}

extern "C" void kernel_launch(void* const* d_in, const int* in_sizes, int n_in,
                              void* d_out, int out_size, void* d_ws, size_t ws_size,
                              hipStream_t stream) {
    (void)in_sizes; (void)n_in; (void)out_size; (void)d_ws; (void)ws_size;
    const float* x = (const float*)d_in[0];
    const float* W = (const float*)d_in[1];
    const float* b = (const float*)d_in[2];
    float* out = (float*)d_out;

    dim3 grid(BDIM / BM, (CDIM + BN - 1) / BN);   // 64 x 8
    gemm_bias_bf16x3_kernel<<<grid, 256, 0, stream>>>(x, W, b, out);
    loo_lse_kernel<<<BDIM, 256, 0, stream>>>(out);
}
